// DepthwiseSeparableConv_71116068487875
// MI455X (gfx1250) — matrix operand order repaired and checked by reading, not compiled, not run on hardware
//
#include <hip/hip_runtime.h>

typedef __attribute__((ext_vector_type(16))) __bf16 v16bf;
typedef __attribute__((ext_vector_type(8)))  __bf16 v8bf;
typedef __attribute__((ext_vector_type(8)))  float  v8f;
typedef __attribute__((ext_vector_type(4), aligned(4))) float f32x4a;  // dword-aligned b128
typedef __attribute__((ext_vector_type(2), aligned(4))) float f32x2a;  // dword-aligned b64

#define BATCH 16
#define CIN   256
#define COUT  512
#define KS    7
#define HH    64
#define WW    64

#define TH 4                 // pixel-tile rows
#define TW 8                 // pixel-tile cols
#define NPIX (TH * TW)       // 32 pixels per image per workgroup
#define NIMG 2               // images per workgroup (N = 64)
#define LDSROW (CIN + 8)     // bf16 elems per LDS row (+16B pad, keeps 16B align)

// ---------------------------------------------------------------------------
// Prep: repack pointwise weights [COUT, CIN] f32 -> bf16 A-fragments laid out
// exactly as V_WMMA_F32_16X16X32_BF16 expects per lane.
// Fragment f = mtile*8 + kstep  (mtile 0..31, kstep 0..7).
// Lane l (0..31): row m = mtile*16 + (l&15);
//   halfs t=0..7  -> K = kstep*32 + (l<16 ? 0 : 8)  + t
//   halfs t=8..15 -> K = kstep*32 + (l<16 ? 16 : 24) + (t-8)
// ---------------------------------------------------------------------------
__global__ __launch_bounds__(256) void pack_pw_kernel(
    const float* __restrict__ pw, __bf16* __restrict__ apack) {
  const int gid  = blockIdx.x * blockDim.x + threadIdx.x;  // 0..8191
  const int lane = gid & 31;
  const int f    = gid >> 5;       // 0..255
  const int kk   = f & 7;
  const int mt   = f >> 3;
  const int m    = mt * 16 + (lane & 15);
  const int kb   = kk * 32 + ((lane & 16) ? 8 : 0);
  v16bf frag;
#pragma unroll
  for (int t = 0; t < 16; ++t) {
    const int k = kb + (t & 7) + ((t & 8) ? 16 : 0);
    frag[t] = (__bf16)pw[(size_t)m * CIN + k];
  }
  *(v16bf*)(apack + (size_t)gid * 16) = frag;
}

// Accumulate one 14-float input row into the 4x8 accumulator block.
// Fully folded after unrolling (r, kh compile-time).
__device__ __forceinline__ void accum_row(int r, const float xrow[TW + KS - 1],
                                          const float wreg[KS][KS],
                                          float acc[TH][TW]) {
#pragma unroll
  for (int kh = 0; kh < KS; ++kh) {
    const int oh = r - kh;
    if (oh >= 0 && oh < TH) {
#pragma unroll
      for (int kw = 0; kw < KS; ++kw)
#pragma unroll
        for (int ow = 0; ow < TW; ++ow)
          acc[oh][ow] += xrow[ow + kw] * wreg[kh][kw];
    }
  }
}

// ---------------------------------------------------------------------------
// Fused depthwise(7x7, f32) -> bf16 LDS -> pointwise GEMM via WMMA bf16.
// Grid: (128 pixel tiles, 8 image pairs). Block: 512 threads (16 wave32).
// Workgroup covers the same 4x8 spatial tile in TWO images (N=64): halves
// pointwise-weight L2 traffic vs N=32. 16 waves keep per-wave accumulators
// at 64 VGPRs (M=32 x N=64 per wave) -> no spills.
// Phase 1: thread = (channel, image); each runs the stencil exactly once.
// ---------------------------------------------------------------------------
__global__ __launch_bounds__(512) void dwsep_wmma_kernel(
    const float*  __restrict__ x,      // [B, CIN, H, W]
    const float*  __restrict__ dww,    // [CIN, 1, 7, 7]
    const float*  __restrict__ dwb,    // [CIN]
    const __bf16* __restrict__ apack,  // packed pw_w fragments
    const float*  __restrict__ pwb,    // [COUT]
    float*        __restrict__ out) {  // [B, COUT, H, W]
  __shared__ __bf16 ylds[NIMG * NPIX][LDSROW];

  const int tid  = threadIdx.x;
  const int b0   = blockIdx.y * NIMG;      // first image of the pair
  const int tile = blockIdx.x;             // 0..127
  const int h0   = (tile >> 3) * TH;       // 16 tiles along H
  const int w0   = (tile & 7)  * TW;       // 8 tiles along W

  // ---------------- Phase 1: depthwise, one (channel, image) per thread -------
  {
    const int c   = tid & (CIN - 1);
    const int img = tid >> 8;              // 0 or 1
    const float* __restrict__ xc =
        x + (((size_t)(b0 + img) * CIN + c) * (HH * WW));

    float wreg[KS][KS];
#pragma unroll
    for (int kh = 0; kh < KS; ++kh)
#pragma unroll
      for (int kw = 0; kw < KS; ++kw)
        wreg[kh][kw] = dww[c * (KS * KS) + kh * KS + kw];
    const float bias = dwb[c];

    float acc[TH][TW];
#pragma unroll
    for (int i = 0; i < TH; ++i)
#pragma unroll
      for (int j = 0; j < TW; ++j) acc[i][j] = bias;

    const bool interior = (h0 >= 3) && (h0 + TH + 3 <= HH) &&
                          (w0 >= 3) && (w0 + TW + 3 <= WW);
    if (interior) {
      // Fast path: no bounds checks, vectorized b128/b64 row loads.
#pragma unroll
      for (int r = 0; r < TH + KS - 1; ++r) {
        const float* __restrict__ rp = xc + (h0 - 3 + r) * WW + (w0 - 3);
        const f32x4a a0 = *(const f32x4a*)(rp);
        const f32x4a a1 = *(const f32x4a*)(rp + 4);
        const f32x4a a2 = *(const f32x4a*)(rp + 8);
        const f32x2a a3 = *(const f32x2a*)(rp + 12);
        float xrow[TW + KS - 1];
#pragma unroll
        for (int t = 0; t < 4; ++t) {
          xrow[t] = a0[t]; xrow[4 + t] = a1[t]; xrow[8 + t] = a2[t];
        }
        xrow[12] = a3[0];
        xrow[13] = a3[1];
        accum_row(r, xrow, wreg, acc);
      }
    } else {
      // Border path: per-element zero-padding.
#pragma unroll
      for (int r = 0; r < TH + KS - 1; ++r) {
        const int ih = h0 - 3 + r;
        float xrow[TW + KS - 1];
        if (ih >= 0 && ih < HH) {
#pragma unroll
          for (int j = 0; j < TW + KS - 1; ++j) {
            const int iw = w0 - 3 + j;
            xrow[j] = (iw >= 0 && iw < WW) ? xc[ih * WW + iw] : 0.0f;
          }
        } else {
#pragma unroll
          for (int j = 0; j < TW + KS - 1; ++j) xrow[j] = 0.0f;
        }
        accum_row(r, xrow, wreg, acc);
      }
    }

    // y[pixel][channel] in bf16 (GEMM B operand is channel-major per pixel)
#pragma unroll
    for (int i = 0; i < TH; ++i)
#pragma unroll
      for (int j = 0; j < TW; ++j)
        ylds[img * NPIX + i * TW + j][c] = (__bf16)acc[i][j];
  }
  __syncthreads();

  // ---------------- Phase 2: pointwise GEMM via WMMA bf16 ----------------
  // Wave w (0..15) owns M rows [w*32, w*32+32) x all 64 pixels: 2x4 tiles.
  const int wave  = tid >> 5;
  const int lane  = tid & 31;
  const int l15   = lane & 15;
  const int lhi   = (lane >> 4) & 1;
  const int mbase = wave * 32;

  v8f accv[2][4];
#pragma unroll
  for (int i2 = 0; i2 < 2; ++i2) {
#pragma unroll
    for (int v = 0; v < 8; ++v) {
      const float bv = pwb[mbase + i2 * 16 + v + lhi * 8];
#pragma unroll
      for (int j = 0; j < 4; ++j) accv[i2][j][v] = bv;
    }
  }

#pragma unroll
  for (int kk = 0; kk < 8; ++kk) {
    // B fragments: 32x16 bf16; K rows striped across lanes: lanes 0-15 hold
    // K = kk*32 + 0..7,16..23, lanes 16-31 hold K = kk*32 + 8..15,24..31
    // (two 8-half chunks per lane -> two ds_load_b128).
    v16bf bfrag[4];
#pragma unroll
    for (int j = 0; j < 4; ++j) {
      const int p  = j * 16 + l15;
      const int kb = kk * 32 + lhi * 8;
      union { v16bf v; v8bf h[2]; } u;
      u.h[0] = *(const v8bf*)&ylds[p][kb];
      u.h[1] = *(const v8bf*)&ylds[p][kb + 16];
      bfrag[j] = u.v;
    }
#pragma unroll
    for (int i2 = 0; i2 < 2; ++i2) {
      const int mt = (mbase >> 4) + i2;
      const v16bf afrag =
          *(const v16bf*)(apack + (((size_t)(mt * 8 + kk)) * 32 + lane) * 16);
#pragma unroll
      for (int j = 0; j < 4; ++j) {
        accv[i2][j] = __builtin_amdgcn_wmma_f32_16x16x32_bf16(
            false, afrag, false, bfrag[j], (short)0, accv[i2][j], false, false);
      }
    }
  }

  // ---------------- Store: C layout -> NCHW f32 ----------------
#pragma unroll
  for (int i2 = 0; i2 < 2; ++i2) {
#pragma unroll
    for (int j = 0; j < 4; ++j) {
      const int p   = j * 16 + l15;          // 0..63
      const int img = p >> 5;                // uniform per j
      const int pp  = p & 31;
      const int oh  = h0 + (pp >> 3);
      const int ow  = w0 + (pp & 7);
      float* __restrict__ outb = out + (size_t)(b0 + img) * COUT * HH * WW;
#pragma unroll
      for (int v = 0; v < 8; ++v) {
        const int m = mbase + i2 * 16 + v + lhi * 8;
        outb[((size_t)m * HH + oh) * WW + ow] = accv[i2][j][v];
      }
    }
  }
}

extern "C" void kernel_launch(void* const* d_in, const int* in_sizes, int n_in,
                              void* d_out, int out_size, void* d_ws, size_t ws_size,
                              hipStream_t stream) {
  (void)in_sizes; (void)n_in; (void)out_size; (void)ws_size;
  const float* x   = (const float*)d_in[0];
  const float* dww = (const float*)d_in[1];
  const float* dwb = (const float*)d_in[2];
  const float* pww = (const float*)d_in[3];
  const float* pwb = (const float*)d_in[4];
  float*  out   = (float*)d_out;
  __bf16* apack = (__bf16*)d_ws;  // 512*256*2 = 256 KB

  pack_pw_kernel<<<dim3(32), dim3(256), 0, stream>>>(pww, apack);
  dwsep_wmma_kernel<<<dim3(128, BATCH / NIMG), dim3(512), 0, stream>>>(
      x, dww, dwb, apack, pwb, out);
}
